// Net_69415261438110
// MI455X (gfx1250) — matrix operand order repaired and checked by reading, not compiled, not run on hardware
//
#include <hip/hip_runtime.h>
#include <hip/hip_bf16.h>
#include <math.h>

// ---------------------------------------------------------------------------
// Types for CDNA5 WMMA (wave32): 16x16x32 bf16 -> f32
// ---------------------------------------------------------------------------
typedef __attribute__((ext_vector_type(16))) __bf16  v16bf;
typedef __attribute__((ext_vector_type(8)))  float   v8f;
typedef __attribute__((ext_vector_type(4)))  unsigned int u32x4;
typedef unsigned int uint32x4 __attribute__((ext_vector_type(4)));
typedef int          int32x8  __attribute__((ext_vector_type(8)));
typedef int          int32x4  __attribute__((ext_vector_type(4)));

union FragBF { v16bf bf; u32x4 q[2]; unsigned short h[16]; };

__device__ __forceinline__ unsigned short f2bf(float f) {
  unsigned int u = __float_as_uint(f);
  u += 0x7FFFu + ((u >> 16) & 1u);          // round-to-nearest-even
  return (unsigned short)(u >> 16);
}

// ---------------------------------------------------------------------------
// Tensor Data Mover: DMA a 64-row x 32-element (bf16) tile, row stride
// `stride_e` elements, from global to LDS. D# packed per CDNA5 ISA 8.3/8.4.
// ---------------------------------------------------------------------------
#if __has_builtin(__builtin_amdgcn_tensor_load_to_lds) && \
    __has_builtin(__builtin_amdgcn_s_wait_tensorcnt)
#define USE_TDM 1
#else
#define USE_TDM 0
#endif

#if USE_TDM
__device__ __forceinline__ void tdm_load_tile64x32(const unsigned short* gptr,
                                                   unsigned lds_off,
                                                   int stride_e) {
  unsigned long long ga = (unsigned long long)(size_t)gptr;
  uint32x4 g0 = {
      1u,                                        // count=1, user desc
      lds_off,                                   // lds_addr (bytes)
      (unsigned)(ga & 0xFFFFFFFFu),              // global_addr[31:0]
      (unsigned)((ga >> 32) & 0x1FFFFFFu) | (2u << 30)  // ga[56:32] | type=2
  };
  int32x8 g1 = {
      (int)(1u << 16),                 // workgroup_mask=0, data_size=1 (2B)
      (int)(32u << 16),                // tensor_dim0[15:0]=32 in w1[31:16]
      (int)(64u << 16),                // td0 hi=0 | tensor_dim1 lo=64
      (int)(32u << 16),                // td1 hi=0 | tile_dim0=32
      (int)64,                         // tile_dim1=64, tile_dim2=0
      (int)stride_e,                   // tensor_dim0_stride[31:0]
      0,                               // stride hi / dim1_stride lo
      0
  };
  int32x4 g2 = {0, 0, 0, 0};
  int32x4 g3 = {0, 0, 0, 0};
#if defined(__clang_major__) && (__clang_major__ >= 23)
  int32x8 g4 = {0, 0, 0, 0, 0, 0, 0, 0};
  __builtin_amdgcn_tensor_load_to_lds(g0, g1, g2, g3, g4, 0);
#else
  __builtin_amdgcn_tensor_load_to_lds(g0, g1, g2, g3, 0);
#endif
}
#endif

// ---------------------------------------------------------------------------
// Tiled GEMM: C[M,N] = A[M,K] @ B[K,N]; bf16 in (u16), f32 out.
// ALL dims/pitches padded: M,N multiples of 64 (grid), K multiple of 32,
// lda/ldb/ldc multiples of 8 -> every access aligned, zero bounds checks.
// 256 threads = 8 waves; 64x64 tile; wave owns col-tile (w&3), row-tiles
// (w>>2) and (w>>2)+2. A tile staged by TDM (wave 0), B tile staged
// transposed so fragments are two ds_load_b128 each (ISA 16-bit layouts).
// ---------------------------------------------------------------------------
__global__ __launch_bounds__(256) void gemm_bf16_wmma(
    const unsigned short* __restrict__ A, const unsigned short* __restrict__ B,
    float* __restrict__ C, int K, int lda, int ldb, int ldc)
{
  __shared__ unsigned short lA[64][32];    // A tile, row major
  __shared__ unsigned short lBT[64][32];   // B tile, transposed (col, k)

  const int t    = threadIdx.x;
  const int wave = t >> 5;
  const int lane = t & 31;
  const int rowBase = blockIdx.y << 6;
  const int colBase = blockIdx.x << 6;

  const int tc  = wave & 3;
  const int trB = wave >> 2;
  const int r   = lane & 15;
  const int hi  = lane >> 4;

  const int kr = t >> 3;            // 0..31  (B tile row)
  const int c0 = (t & 7) << 3;      // 0..56  (B tile col chunk)

  v8f acc0 = {};
  v8f acc1 = {};

  for (int k0 = 0; k0 < K; k0 += 32) {
#if USE_TDM
    if (wave == 0) {
      tdm_load_tile64x32(A + (size_t)rowBase * lda + k0,
                         (unsigned)(size_t)(void*)&lA[0][0], lda);
      __builtin_amdgcn_s_wait_tensorcnt(0);
    }
#else
    {
      int ar = t >> 2;                    // 0..63
      int ck = (t & 3) << 3;              // 0,8,16,24
      *(u32x4*)&lA[ar][ck] =
          *(const u32x4*)(A + (size_t)(rowBase + ar) * lda + (k0 + ck));
    }
#endif
    {
      union { u32x4 q; unsigned short s[8]; } u;
      u.q = *(const u32x4*)(B + (size_t)(k0 + kr) * ldb + (colBase + c0));
      #pragma unroll
      for (int j = 0; j < 8; ++j) lBT[c0 + j][kr] = u.s[j];
    }
    __syncthreads();

    FragBF b, a0, a1;
    b.q[0]  = *(const u32x4*)&lBT[tc * 16 + r][hi * 8];
    b.q[1]  = *(const u32x4*)&lBT[tc * 16 + r][16 + hi * 8];
    a0.q[0] = *(const u32x4*)&lA[trB * 16 + r][hi * 8];
    a0.q[1] = *(const u32x4*)&lA[trB * 16 + r][16 + hi * 8];
    a1.q[0] = *(const u32x4*)&lA[(trB + 2) * 16 + r][hi * 8];
    a1.q[1] = *(const u32x4*)&lA[(trB + 2) * 16 + r][16 + hi * 8];

    acc0 = __builtin_amdgcn_wmma_f32_16x16x32_bf16(
        false, a0.bf, false, b.bf, (short)0, acc0, false, false);
    acc1 = __builtin_amdgcn_wmma_f32_16x16x32_bf16(
        false, a1.bf, false, b.bf, (short)0, acc1, false, false);

    __syncthreads();
  }

  const int col = colBase + tc * 16 + r;
  #pragma unroll
  for (int vv = 0; vv < 8; ++vv) {
    C[(size_t)(rowBase + trB * 16 + vv + hi * 8) * ldc + col]       = acc0[vv];
    C[(size_t)(rowBase + (trB + 2) * 16 + vv + hi * 8) * ldc + col] = acc1[vv];
  }
}

// ---------------------------------------------------------------------------
// Elementwise / helper kernels (deterministic, no atomics)
// ---------------------------------------------------------------------------
__global__ void fill_f32(float* __restrict__ p, float v, int n) {
  int i = blockIdx.x * blockDim.x + threadIdx.x;
  if (i < n) p[i] = v;
}

__global__ void copy_f32(const float* __restrict__ s, float* __restrict__ d, int n) {
  int i = blockIdx.x * blockDim.x + threadIdx.x;
  if (i < n) d[i] = s[i];
}

__global__ void scatter_edges(const int* __restrict__ ei, float* __restrict__ A,
                              int E, int N) {
  int t = blockIdx.x * blockDim.x + threadIdx.x;
  if (t >= E) return;
  int s = ei[t], d = ei[E + t];
  if (s != d) A[(size_t)s * N + d] = 1.0f;
}

// dst[r*dpitch+c] = bf16(src[r*spitch+c]) inside srows x scols, else 0.
// grid: ((dcols+255)/256, drows)
__global__ void cvt_pad_bf16(const float* __restrict__ src, int srows, int scols,
                             int spitch, unsigned short* __restrict__ dst,
                             int dcols, int dpitch) {
  int c = blockIdx.x * blockDim.x + threadIdx.x;
  int rr = blockIdx.y;
  if (c >= dcols) return;
  float v = (rr < srows && c < scols) ? src[(size_t)rr * spitch + c] : 0.0f;
  dst[(size_t)rr * dpitch + c] = f2bf(v);
}

// bf16(A + I) over padded P x P (identity only inside n x n). grid ((P+255)/256, P)
__global__ void addI_pad_bf16(const float* __restrict__ A, int n, int P,
                              unsigned short* __restrict__ out) {
  int c = blockIdx.x * blockDim.x + threadIdx.x;
  int rr = blockIdx.y;
  if (c >= P) return;
  float v = (rr < n && c < n) ? (A[(size_t)rr * P + c] + (rr == c ? 1.0f : 0.0f))
                              : 0.0f;
  out[(size_t)rr * P + c] = f2bf(v);
}

// out[r,c] = B[perm[r], perm[c]] inside k x k, else 0. grid ((Pout+255)/256, Pout)
__global__ void gather_A_pad(const float* __restrict__ B, int Pin,
                             const int* __restrict__ perm,
                             float* __restrict__ out, int k, int Pout) {
  int c = blockIdx.x * blockDim.x + threadIdx.x;
  int rr = blockIdx.y;
  if (c >= Pout) return;
  float v = 0.0f;
  if (rr < k && c < k) v = B[(size_t)perm[rr] * Pin + perm[c]];
  out[(size_t)rr * Pout + c] = v;
}

__global__ void zero_diag(float* __restrict__ A, int n, int P) {
  int i = blockIdx.x * blockDim.x + threadIdx.x;
  if (i < n) A[(size_t)i * P + i] = 0.0f;
}

// dinv[row] = (rowsum(A) + 2)^-1/2 over pitch-P rows (pads are zero)
__global__ void row_deg_dinv(const float* __restrict__ A,
                             float* __restrict__ dinv, int P) {
  __shared__ float red[256];
  int row = blockIdx.x;
  float s = 0.f;
  for (int j = threadIdx.x; j < P; j += 256) s += A[(size_t)row * P + j];
  red[threadIdx.x] = s;
  __syncthreads();
  for (int off = 128; off > 0; off >>= 1) {
    if (threadIdx.x < off) red[threadIdx.x] += red[threadIdx.x + off];
    __syncthreads();
  }
  if (threadIdx.x == 0) {
    float deg = red[0] + 2.0f;
    dinv[row] = (deg > 0.f) ? rsqrtf(deg) : 0.f;
  }
}

// hd_bf[r,c] = bf16(h[r,c]*dinv[r]) for r<n, else 0. grid (1, P), block 64
__global__ void scale_rows_pad_bf16(const float* __restrict__ h,
                                    const float* __restrict__ dinv,
                                    unsigned short* __restrict__ out, int n) {
  int c = threadIdx.x;           // 0..63
  int rr = blockIdx.y;
  float v = (rr < n) ? h[(size_t)rr * 64 + c] * dinv[rr] : 0.0f;
  out[(size_t)rr * 64 + c] = f2bf(v);
}

// out[r*outC+c] = dinv*y + 2*dinv^2*h + b ; optional relu. grid (1, n), block 64
__global__ void gcn_epilogue(const float* __restrict__ y,
                             const float* __restrict__ h,
                             const float* __restrict__ dinv,
                             const float* __restrict__ b,
                             float* __restrict__ out,
                             int outC, int do_relu) {
  int c = threadIdx.x;
  int rr = blockIdx.y;
  if (c >= outC) return;
  float d = dinv[rr];
  float v = d * y[(size_t)rr * 64 + c] + 2.0f * d * d * h[(size_t)rr * 64 + c] + b[c];
  if (do_relu) v = fmaxf(v, 0.0f);
  out[(size_t)rr * outC + c] = v;
}

__global__ void pool_score(const float* __restrict__ x, const float* __restrict__ p,
                           float* __restrict__ score, int rows, int cols) {
  int i = blockIdx.x * blockDim.x + threadIdx.x;
  if (i >= rows) return;
  float dot = 0.f, pn = 0.f;
  for (int j = 0; j < cols; ++j) {
    float pv = p[j];
    dot += x[(size_t)i * cols + j] * pv;
    pn  += pv * pv;
  }
  score[i] = tanhf(dot * rsqrtf(pn));
}

// stable top-k by rank counting (matches lax.top_k ordering & tie-break)
__global__ void topk_perm(const float* __restrict__ score, int* __restrict__ perm,
                          int n, int k) {
  int i = blockIdx.x * blockDim.x + threadIdx.x;
  if (i >= n) return;
  float si = score[i];
  int cnt = 0;
  for (int j = 0; j < n; ++j) {
    float sj = score[j];
    cnt += (sj > si) || (sj == si && j < i);
  }
  if (cnt < k) perm[cnt] = i;
}

__global__ void gather_x(const float* __restrict__ x, const float* __restrict__ score,
                         const int* __restrict__ perm, float* __restrict__ out,
                         int k, int cols) {
  int idx = blockIdx.x * blockDim.x + threadIdx.x;
  if (idx >= k * cols) return;
  int rr = idx / cols, c = idx - rr * cols;
  int p = perm[rr];
  out[idx] = x[(size_t)p * cols + c] * score[p];
}

// out[perm[r], c] += x[r, c]   (perm unique -> no atomics)
__global__ void scatter_add(const float* __restrict__ x, const int* __restrict__ perm,
                            float* __restrict__ out, int k, int cols) {
  int idx = blockIdx.x * blockDim.x + threadIdx.x;
  if (idx >= k * cols) return;
  int rr = idx / cols, c = idx - rr * cols;
  out[(size_t)perm[rr] * cols + c] += x[idx];
}

// tanh -> 2x16 -> tanh -> 16x32 -> tanh -> 32x2
__global__ void head_mlp(const float* __restrict__ hin,
                         const float* __restrict__ l1W, const float* __restrict__ l1b,
                         const float* __restrict__ l2W, const float* __restrict__ l2b,
                         const float* __restrict__ fW,  const float* __restrict__ fb,
                         float* __restrict__ out, int n) {
  int i = blockIdx.x * blockDim.x + threadIdx.x;
  if (i >= n) return;
  float a = tanhf(hin[2 * i]), b = tanhf(hin[2 * i + 1]);
  float h1[16];
  #pragma unroll
  for (int j = 0; j < 16; ++j) h1[j] = tanhf(a * l1W[j] + b * l1W[16 + j] + l1b[j]);
  float h2[32];
  for (int j = 0; j < 32; ++j) {
    float s = l2b[j];
    #pragma unroll
    for (int q = 0; q < 16; ++q) s += h1[q] * l2W[q * 32 + j];
    h2[j] = tanhf(s);
  }
  float o0 = fb[0], o1 = fb[1];
  #pragma unroll
  for (int q = 0; q < 32; ++q) { o0 += h2[q] * fW[q * 2]; o1 += h2[q] * fW[q * 2 + 1]; }
  out[2 * i] = o0; out[2 * i + 1] = o1;
}

// fixed-order column Gram matrix of [n,2]
__global__ void qr_reduce(const float* __restrict__ h, float* __restrict__ s, int n) {
  __shared__ float r0[256], r1[256], r2[256];
  float s00 = 0.f, s01 = 0.f, s11 = 0.f;
  for (int i = threadIdx.x; i < n; i += 256) {
    float a = h[2 * i], b = h[2 * i + 1];
    s00 += a * a; s01 += a * b; s11 += b * b;
  }
  r0[threadIdx.x] = s00; r1[threadIdx.x] = s01; r2[threadIdx.x] = s11;
  __syncthreads();
  for (int off = 128; off > 0; off >>= 1) {
    if (threadIdx.x < off) {
      r0[threadIdx.x] += r0[threadIdx.x + off];
      r1[threadIdx.x] += r1[threadIdx.x + off];
      r2[threadIdx.x] += r2[threadIdx.x + off];
    }
    __syncthreads();
  }
  if (threadIdx.x == 0) { s[0] = r0[0]; s[1] = r1[0]; s[2] = r2[0]; }
}

// Gram-Schmidt Q from [n,2] using precomputed Gram entries
__global__ void qr_write(const float* __restrict__ h, const float* __restrict__ s,
                         float* __restrict__ q, int n) {
  int i = blockIdx.x * blockDim.x + threadIdx.x;
  if (i >= n) return;
  float s00 = s[0], s01 = s[1], s11 = s[2];
  float r00 = sqrtf(s00);
  float inv00 = (r00 > 0.f) ? 1.0f / r00 : 0.f;
  float a = h[2 * i], b = h[2 * i + 1];
  float q1 = a * inv00;
  float proj = s01 * inv00;
  float v = b - proj * q1;
  float r11sq = s11 - proj * proj;
  float inv11 = (r11sq > 0.f) ? rsqrtf(r11sq) : 0.f;
  q[2 * i] = q1;
  q[2 * i + 1] = v * inv11;
}

// ---------------------------------------------------------------------------
// Host orchestration
// ---------------------------------------------------------------------------
extern "C" void kernel_launch(void* const* d_in, const int* in_sizes, int n_in,
                              void* d_out, int out_size, void* d_ws, size_t ws_size,
                              hipStream_t stream) {
  (void)in_sizes; (void)out_size; (void)ws_size;
  if (n_in < 25) return;

  const int N = 4096, E = 65536;
  const int k1 = 3892, k2 = 3309, k3 = 2813;       // ceil-topk sizes
  const int P0 = 4096, P1 = 3904, P2 = 3328, P3 = 2816;  // padded to 64

  // ---- inputs (setup_inputs order; params flattened in insertion order) ----
  const float* xin  = (const float*)d_in[0];
  const int*   ei   = (const int*)d_in[1];
  const float* dW[4] = {(const float*)d_in[2], (const float*)d_in[3],
                        (const float*)d_in[4], (const float*)d_in[5]};
  const float* db[4] = {(const float*)d_in[6], (const float*)d_in[7],
                        (const float*)d_in[8], (const float*)d_in[9]};
  const float* pp[3] = {(const float*)d_in[10], (const float*)d_in[11],
                        (const float*)d_in[12]};
  const float* uW[3] = {(const float*)d_in[13], (const float*)d_in[14],
                        (const float*)d_in[15]};
  const float* ub[3] = {(const float*)d_in[16], (const float*)d_in[17],
                        (const float*)d_in[18]};
  const float* l1W = (const float*)d_in[19];
  const float* l1b = (const float*)d_in[20];
  const float* l2W = (const float*)d_in[21];
  const float* l2b = (const float*)d_in[22];
  const float* fW  = (const float*)d_in[23];
  const float* fb  = (const float*)d_in[24];

  // ---- workspace carve ----
  char* cur = (char*)d_ws;
  auto wsalloc = [&](size_t bytes) -> void* {
    void* p = (void*)cur;
    cur += (bytes + 255) & ~(size_t)255;
    return p;
  };
  const size_t NN = (size_t)P0 * P0;
  float* A0   = (float*)wsalloc(NN * 4);              // pitch P0
  float* A1s  = (float*)wsalloc((size_t)P1 * P1 * 4); // pitch P1
  float* A2s  = (float*)wsalloc((size_t)P2 * P2 * 4); // pitch P2
  float* A3s  = (float*)wsalloc((size_t)P3 * P3 * 4); // pitch P3
  float* Btmp = (float*)wsalloc(NN * 4);              // augment output, pitch Pi
  unsigned short* Abf = (unsigned short*)wsalloc(NN * 2);
  unsigned short* xbf = (unsigned short*)wsalloc((size_t)P0 * 64 * 2);
  unsigned short* wbf = (unsigned short*)wsalloc(64 * 64 * 2);
  float* hbuf = (float*)wsalloc((size_t)P0 * 64 * 4);
  float* ybuf = (float*)wsalloc((size_t)P0 * 64 * 4);
  float* xcur = (float*)wsalloc((size_t)N * 64 * 4);
  float* xtmp = (float*)wsalloc((size_t)N * 64 * 4);
  float* x0s  = (float*)wsalloc((size_t)N * 64 * 4);
  float* x1s  = (float*)wsalloc((size_t)N * 64 * 4);
  float* x2s  = (float*)wsalloc((size_t)N * 64 * 4);
  float* dinv  = (float*)wsalloc(N * 4);
  float* score = (float*)wsalloc(N * 4);
  int* perm0 = (int*)wsalloc(N * 4);
  int* perm1 = (int*)wsalloc(N * 4);
  int* perm2 = (int*)wsalloc(N * 4);
  float* sums = (float*)wsalloc(256);

  auto g1 = [](int n) { return dim3((unsigned)((n + 255) / 256)); };
  auto g2 = [](int cols, int rows) {
    return dim3((unsigned)((cols + 255) / 256), (unsigned)rows);
  };
  auto gemm = [&](const unsigned short* A, const unsigned short* B, float* C,
                  int M, int Nc, int K, int lda, int ldb, int ldc) {
    gemm_bf16_wmma<<<dim3((unsigned)(Nc / 64), (unsigned)(M / 64)),
                     256, 0, stream>>>(A, B, C, K, lda, ldb, ldc);
  };

  // GCN: out = d⊙(A@(d⊙(xW))) + 2d²⊙(xW) + b  [optional relu]; A pitch P
  auto gcn = [&](const float* Aptr, int n, int P, const float* xp, int inC,
                 const float* W, const float* b, int outC, float* out, int relu) {
    cvt_pad_bf16<<<g2(inC, P), 256, 0, stream>>>(xp, n, inC, inC, xbf, inC, inC);
    cvt_pad_bf16<<<g2(64, 64), 256, 0, stream>>>(W, inC, outC, outC, wbf, 64, 64);
    gemm(xbf, wbf, hbuf, P, 64, inC, inC, 64, 64);        // h = x @ W
    row_deg_dinv<<<n, 256, 0, stream>>>(Aptr, dinv, P);
    scale_rows_pad_bf16<<<dim3(1, P), 64, 0, stream>>>(hbuf, dinv, xbf, n);
    cvt_pad_bf16<<<g2(P, P), 256, 0, stream>>>(Aptr, P, P, P, Abf, P, P);
    gemm(Abf, xbf, ybuf, P, 64, P, P, 64, 64);            // y = A @ (d⊙h)
    gcn_epilogue<<<dim3(1, n), 64, 0, stream>>>(ybuf, hbuf, dinv, b, out,
                                                outC, relu);
  };

  // augment: Aout = (A+I)@(A+I), zero diagonal; pitch P preserved
  auto augment = [&](const float* Ain, int n, int P, float* Aout) {
    addI_pad_bf16<<<g2(P, P), 256, 0, stream>>>(Ain, n, P, Abf);
    gemm(Abf, Abf, Aout, P, P, P, P, P, P);
    zero_diag<<<g1(n), 256, 0, stream>>>(Aout, n, P);
  };

  // ---- build dense adjacency (P0 == N, no pads) ----
  fill_f32<<<g1((int)NN), 256, 0, stream>>>(A0, 0.0f, (int)NN);
  scatter_edges<<<g1(E), 256, 0, stream>>>(ei, A0, E, N);

  // ---- down path ----
  gcn(A0, N, P0, xin, 32, dW[0], db[0], 64, xcur, 1);
  copy_f32<<<g1(N * 64), 256, 0, stream>>>(xcur, x0s, N * 64);

  // level 1
  augment(A0, N, P0, Btmp);
  pool_score<<<g1(N), 256, 0, stream>>>(xcur, pp[0], score, N, 64);
  topk_perm<<<g1(N), 256, 0, stream>>>(score, perm0, N, k1);
  gather_x<<<g1(k1 * 64), 256, 0, stream>>>(xcur, score, perm0, xtmp, k1, 64);
  gather_A_pad<<<g2(P1, P1), 256, 0, stream>>>(Btmp, P0, perm0, A1s, k1, P1);
  gcn(A1s, k1, P1, xtmp, 64, dW[1], db[1], 64, xcur, 1);
  copy_f32<<<g1(k1 * 64), 256, 0, stream>>>(xcur, x1s, k1 * 64);

  // level 2
  augment(A1s, k1, P1, Btmp);
  pool_score<<<g1(k1), 256, 0, stream>>>(xcur, pp[1], score, k1, 64);
  topk_perm<<<g1(k1), 256, 0, stream>>>(score, perm1, k1, k2);
  gather_x<<<g1(k2 * 64), 256, 0, stream>>>(xcur, score, perm1, xtmp, k2, 64);
  gather_A_pad<<<g2(P2, P2), 256, 0, stream>>>(Btmp, P1, perm1, A2s, k2, P2);
  gcn(A2s, k2, P2, xtmp, 64, dW[2], db[2], 64, xcur, 1);
  copy_f32<<<g1(k2 * 64), 256, 0, stream>>>(xcur, x2s, k2 * 64);

  // level 3
  augment(A2s, k2, P2, Btmp);
  pool_score<<<g1(k2), 256, 0, stream>>>(xcur, pp[2], score, k2, 64);
  topk_perm<<<g1(k2), 256, 0, stream>>>(score, perm2, k2, k3);
  gather_x<<<g1(k3 * 64), 256, 0, stream>>>(xcur, score, perm2, xtmp, k3, 64);
  gather_A_pad<<<g2(P3, P3), 256, 0, stream>>>(Btmp, P2, perm2, A3s, k3, P3);
  gcn(A3s, k3, P3, xtmp, 64, dW[3], db[3], 64, xcur, 1);

  // ---- up path (sum_res skip connections) ----
  copy_f32<<<g1(k2 * 64), 256, 0, stream>>>(x2s, xtmp, k2 * 64);
  scatter_add<<<g1(k3 * 64), 256, 0, stream>>>(xcur, perm2, xtmp, k3, 64);
  gcn(A2s, k2, P2, xtmp, 64, uW[0], ub[0], 64, xcur, 1);

  copy_f32<<<g1(k1 * 64), 256, 0, stream>>>(x1s, xtmp, k1 * 64);
  scatter_add<<<g1(k2 * 64), 256, 0, stream>>>(xcur, perm1, xtmp, k2, 64);
  gcn(A1s, k1, P1, xtmp, 64, uW[1], ub[1], 64, xcur, 1);

  copy_f32<<<g1(N * 64), 256, 0, stream>>>(x0s, xtmp, N * 64);
  scatter_add<<<g1(k1 * 64), 256, 0, stream>>>(xcur, perm0, xtmp, k1, 64);
  gcn(A0, N, P0, xtmp, 64, uW[2], ub[2], 2, xcur, 0);     // -> [N,2]

  // ---- head MLP + QR ----
  head_mlp<<<g1(N), 256, 0, stream>>>(xcur, l1W, l1b, l2W, l2b, fW, fb, xtmp, N);
  qr_reduce<<<1, 256, 0, stream>>>(xtmp, sums, N);
  qr_write<<<g1(N), 256, 0, stream>>>(xtmp, sums, (float*)d_out, N);
}
